// BitLTIInjection_88321707475629
// MI455X (gfx1250) — 8-bit matrix operand order repaired and checked by reading, compiled, not run on hardware
//
#include <hip/hip_runtime.h>
#include <math.h>

typedef __attribute__((ext_vector_type(8))) int v8i;

#define EPSQ    1e-5f
#define D_DIM   2048
#define NTOK    16384
#define WS_SUM  0
#define WS_ACT  1024
#define WS_WQ   131072
#define WS_EQ   (131072 + 4194304)

__global__ void k_absw(const float* __restrict__ W, float* __restrict__ sum) {
    __shared__ float red[256];
    float acc = 0.f;
    int stride = gridDim.x * blockDim.x;
    for (int i = blockIdx.x * blockDim.x + threadIdx.x; i < D_DIM * D_DIM; i += stride)
        acc += fabsf(W[i]);
    red[threadIdx.x] = acc;
    __syncthreads();
    for (int s = 128; s > 0; s >>= 1) {
        if (threadIdx.x < (unsigned)s) red[threadIdx.x] += red[threadIdx.x + s];
        __syncthreads();
    }
    if (threadIdx.x == 0) atomicAdd(sum, red[0]);
}

__global__ void k_quant_w(const float* __restrict__ W, const float* __restrict__ sum,
                          unsigned char* __restrict__ wq) {
    int idx = blockIdx.x * blockDim.x + threadIdx.x;
    int o  = idx >> 9;
    int d0 = (idx & 511) << 2;
    float mean = *sum * (1.0f / (float)(D_DIM * D_DIM));
    float sw = 1.0f / fmaxf(mean, EPSQ);
    unsigned int pack = 0;
#pragma unroll
    for (int i = 0; i < 4; ++i) {
        float q = rintf(W[(size_t)o * D_DIM + d0 + i] * sw);
        q = fminf(fmaxf(q, -1.f), 1.f);
        pack |= ((unsigned int)((int)q & 0xFF)) << (8 * i);
    }
    int nb = o >> 4, n = o & 15;
    int kb = d0 >> 6, kk = d0 & 63;
    int lane = n + 16 * ((kk >> 4) & 1);
    int v    = (kk >> 5) * 4 + ((kk >> 2) & 3);
    *(unsigned int*)(wq + (((size_t)(nb * 32 + kb)) << 10) + lane * 32 + v * 4) = pack;
}

__global__ void k_quant_e(const float* __restrict__ e, unsigned char* __restrict__ eq,
                          float* __restrict__ actInv) {
    __shared__ float red[256];
    int t = blockIdx.x, tid = threadIdx.x;
    const float4* row = (const float4*)(e + (size_t)t * D_DIM);
    float4 x0 = row[tid * 2 + 0];
    float4 x1 = row[tid * 2 + 1];
    float m = fmaxf(fmaxf(fabsf(x0.x), fabsf(x0.y)), fmaxf(fabsf(x0.z), fabsf(x0.w)));
    m = fmaxf(m, fmaxf(fmaxf(fabsf(x1.x), fabsf(x1.y)), fmaxf(fabsf(x1.z), fabsf(x1.w))));
    red[tid] = m;
    __syncthreads();
    for (int s = 128; s > 0; s >>= 1) {
        if (tid < (unsigned)s) red[tid] = fmaxf(red[tid], red[tid + s]);
        __syncthreads();
    }
    float amax  = fmaxf(red[0], EPSQ);
    float scale = 127.0f / amax;
    if (tid == 0) actInv[t] = amax * (1.0f / 127.0f);

    float vals[8] = {x0.x, x0.y, x0.z, x0.w, x1.x, x1.y, x1.z, x1.w};
    unsigned int p0 = 0, p1 = 0;
#pragma unroll
    for (int i = 0; i < 4; ++i) {
        float q = fminf(fmaxf(rintf(vals[i] * scale), -128.f), 127.f);
        p0 |= ((unsigned int)((int)q & 0xFF)) << (8 * i);
    }
#pragma unroll
    for (int i = 0; i < 4; ++i) {
        float q = fminf(fmaxf(rintf(vals[4 + i] * scale), -128.f), 127.f);
        p1 |= ((unsigned int)((int)q & 0xFF)) << (8 * i);
    }
    int d0 = tid * 8;
    int mrow = t >> 4, r = t & 15;
    int kb = d0 >> 6, kk = d0 & 63;
    int lane = r + 16 * ((kk >> 4) & 1);
    int v0   = (kk >> 5) * 4 + ((kk >> 3) & 1) * 2;
    uint2 pp; pp.x = p0; pp.y = p1;
    *(uint2*)(eq + (((size_t)(mrow * 32 + kb)) << 10) + lane * 32 + v0 * 4) = pp;
}

__global__ void __launch_bounds__(256) k_gemm(
    const float* __restrict__ h, const float* __restrict__ bo,
    const float* __restrict__ Araw, const float* __restrict__ sumAbs,
    const float* __restrict__ actInv,
    const unsigned char* __restrict__ eq, const unsigned char* __restrict__ wq,
    float* __restrict__ out)
{
    const int lane   = threadIdx.x & 31;
    const int wave   = threadIdx.x >> 5;
    const int mTile  = blockIdx.x;
    const int nStrip = blockIdx.y * 8 + wave;

    const unsigned char* aBase = eq + (((size_t)mTile * 32) << 10) + lane * 32;
    const unsigned char* bBase = wq + (((size_t)nStrip * 4 * 32) << 10) + lane * 32;

    v8i acc0 = {}, acc1 = {}, acc2 = {}, acc3 = {};
    for (int kb = 0; kb < 32; ++kb) {
        if (kb < 31) {
            __builtin_prefetch(aBase + ((size_t)(kb + 1) << 10), 0, 3);
            __builtin_prefetch(bBase + ((size_t)(kb + 1) << 10), 0, 3);
        }
        v8i a  = *(const v8i*)(aBase + ((size_t)kb << 10));
        v8i b0 = *(const v8i*)(bBase + ((size_t)(0 * 32 + kb) << 10));
        v8i b1 = *(const v8i*)(bBase + ((size_t)(1 * 32 + kb) << 10));
        v8i b2 = *(const v8i*)(bBase + ((size_t)(2 * 32 + kb) << 10));
        v8i b3 = *(const v8i*)(bBase + ((size_t)(3 * 32 + kb) << 10));
        acc0 = __builtin_amdgcn_wmma_i32_16x16x64_iu8(true, a, true, b0, acc0, false, false);
        acc1 = __builtin_amdgcn_wmma_i32_16x16x64_iu8(true, a, true, b1, acc1, false, false);
        acc2 = __builtin_amdgcn_wmma_i32_16x16x64_iu8(true, a, true, b2, acc2, false, false);
        acc3 = __builtin_amdgcn_wmma_i32_16x16x64_iu8(true, a, true, b3, acc3, false, false);
    }

    float mean = *sumAbs * (1.0f / (float)(D_DIM * D_DIM));
    float winv = fmaxf(mean, EPSQ);
    const int hi = lane >> 4, col = lane & 15;
    const int n0 = nStrip * 64;
    float ae0 = 0.99f * tanhf(Araw[n0 +  0 + col]);
    float ae1 = 0.99f * tanhf(Araw[n0 + 16 + col]);
    float ae2 = 0.99f * tanhf(Araw[n0 + 32 + col]);
    float ae3 = 0.99f * tanhf(Araw[n0 + 48 + col]);

#pragma unroll
    for (int v = 0; v < 8; ++v) {
        int tok = mTile * 16 + v + 8 * hi;
        float ps = actInv[tok] * winv;
        size_t rowOff = (size_t)tok * D_DIM;
        size_t i0 = rowOff + n0 + col;
        out[i0]      = ae0 * h[i0]      + (float)acc0[v] * ps + bo[i0];
        out[i0 + 16] = ae1 * h[i0 + 16] + (float)acc1[v] * ps + bo[i0 + 16];
        out[i0 + 32] = ae2 * h[i0 + 32] + (float)acc2[v] * ps + bo[i0 + 32];
        out[i0 + 48] = ae3 * h[i0 + 48] + (float)acc3[v] * ps + bo[i0 + 48];
    }
}

extern "C" void kernel_launch(void* const* d_in, const int* in_sizes, int n_in,
                              void* d_out, int out_size, void* d_ws, size_t ws_size,
                              hipStream_t stream) {
    const float* h    = (const float*)d_in[0];
    const float* e    = (const float*)d_in[1];
    const float* bo   = (const float*)d_in[2];
    const float* Araw = (const float*)d_in[3];
    const float* W    = (const float*)d_in[4];
    float* out = (float*)d_out;

    char* ws = (char*)d_ws;
    float* sum             = (float*)(ws + WS_SUM);
    float* actInv          = (float*)(ws + WS_ACT);
    unsigned char* wq      = (unsigned char*)(ws + WS_WQ);
    unsigned char* eq      = (unsigned char*)(ws + WS_EQ);

    hipMemsetAsync(sum, 0, sizeof(float), stream);
    k_absw   <<<1024, 256, 0, stream>>>(W, sum);
    k_quant_w<<<4096, 256, 0, stream>>>(W, sum, wq);
    k_quant_e<<<NTOK, 256, 0, stream>>>(e, eq, actInv);
    k_gemm   <<<dim3(1024, 4), 256, 0, stream>>>(h, bo, Araw, sum, actInv, eq, wq, out);
}
